// ConvAttention_30039001269052
// MI455X (gfx1250) — matrix operand order repaired and checked by reading, compiled, not run on hardware
//
#include <hip/hip_runtime.h>

typedef __attribute__((ext_vector_type(16))) __bf16 v16bf;
typedef __attribute__((ext_vector_type(8)))  float  v8f;
typedef __attribute__((ext_vector_type(4)))  unsigned tdm_g0_t;
typedef __attribute__((ext_vector_type(8)))  int      tdm_g1_t;
typedef __attribute__((ext_vector_type(4)))  int      tdm_g2_t;
typedef __attribute__((ext_vector_type(8)))  int      tdm_g4_t;

#define CH    512
#define NH    8
#define DH    64
#define NPIX  1024
#define NKV   1028
#define NKVP  1056   // 33 * 32, padded key count
#define O3    1536

#if defined(__has_builtin)
#if __has_builtin(__builtin_amdgcn_tensor_load_to_lds) && __has_builtin(__builtin_amdgcn_s_wait_tensorcnt)
#define USE_TDM 1
#endif
#endif
#ifndef USE_TDM
#define USE_TDM 0
#endif

union BFV { v16bf v; unsigned u[8]; __bf16 h[16]; };

__device__ __forceinline__ unsigned pack_f2bf(float a, float b) {
  union { __bf16 h[2]; unsigned u; } t;
  t.h[0] = (__bf16)a; t.h[1] = (__bf16)b;
  return t.u;
}
__device__ __forceinline__ unsigned pack_bf2(__bf16 a, __bf16 b) {
  union { __bf16 h[2]; unsigned u; } t;
  t.h[0] = a; t.h[1] = b;
  return t.u;
}

// butterfly reductions across the 16 lanes of a half-wave (wave32)
__device__ __forceinline__ float red_max16(float x) {
  x = fmaxf(x, __shfl_xor(x, 1, 32));
  x = fmaxf(x, __shfl_xor(x, 2, 32));
  x = fmaxf(x, __shfl_xor(x, 4, 32));
  x = fmaxf(x, __shfl_xor(x, 8, 32));
  return x;
}
__device__ __forceinline__ float red_sum16(float x) {
  x += __shfl_xor(x, 1, 32);
  x += __shfl_xor(x, 2, 32);
  x += __shfl_xor(x, 4, 32);
  x += __shfl_xor(x, 8, 32);
  return x;
}

// 16-bit A-matrix 16x32 layout (contiguous K): dword j of lane (m, half) holds K = half*16 + 2j, +1
__device__ __forceinline__ int a_kidx(int j, int half) {
  return half * 16 + 2 * j;
}

#if USE_TDM
// Issue a TDM load of a [rows x 64] bf16 tile (row-major, stride 64) into LDS.
// D# per CDNA5 ISA: group0 = {flags, lds_addr,
// global_addr lo, global_addr hi | type=2}; group1 packs data_size/dims/strides.
// 6-arg builtin variant (clang-23 / therock-10.0 headers).
__device__ __forceinline__ void tdm_load_tile_bf16(const __bf16* gsrc, __bf16* lds_dst,
                                                   int rows) {
  unsigned long long ga = (unsigned long long)(const void*)gsrc;
  unsigned lds_off = (unsigned)(unsigned long long)(const void*)lds_dst;
  tdm_g0_t g0;
  g0[0] = 1u;                                   // count=1, user mode
  g0[1] = lds_off;                              // lds_addr
  g0[2] = (unsigned)(ga & 0xffffffffu);         // global_addr[31:0]
  g0[3] = (unsigned)((ga >> 32) & 0x01ffffffu)  // global_addr[56:32]
          | 0x80000000u;                        // type=2 ("image")
  tdm_g1_t g1;
  g1[0] = 0x00010000;                           // wg_mask=0, data_size=1 (2 bytes)
  g1[1] = (int)(64u << 16);                     // tensor_dim0[15:0] -> bits 63:48
  g1[2] = (int)(((unsigned)rows) << 16);        // tensor_dim0 hi=0, tensor_dim1 lo
  g1[3] = (int)(64u << 16);                     // tensor_dim1 hi=0, tile_dim0=64
  g1[4] = rows;                                 // tile_dim1=rows, tile_dim2=0
  g1[5] = 64;                                   // tensor_dim0_stride lo
  g1[6] = 0;                                    // stride0 hi, stride1 lo
  g1[7] = 0;                                    // stride1 hi
  tdm_g2_t gz = {0, 0, 0, 0};                   // 2D tensor: groups 2/3 unused
  tdm_g4_t gz8 = {0, 0, 0, 0, 0, 0, 0, 0};      // unused trailing group
  __builtin_amdgcn_tensor_load_to_lds(g0, g1, gz, gz, gz8, 0);
}
#endif

// ---------------------------------------------------------------------------
// Kernel 1: forced weight normalization -> bf16 weights
// rows 0..1535: w_qkv, rows 1536..2047: w_out.  eff = w / (||w_row|| + eps*sqrt(512))
// ---------------------------------------------------------------------------
__global__ void __launch_bounds__(256)
wnorm_kernel(const float* __restrict__ w_qkv, const float* __restrict__ w_out,
             __bf16* __restrict__ wqn, __bf16* __restrict__ won) {
  __shared__ float red[256];
  int r = blockIdx.x;
  const float* src;
  __bf16* dst;
  if (r < O3) { src = w_qkv + (size_t)r * CH;        dst = wqn + (size_t)r * CH; }
  else        { src = w_out + (size_t)(r - O3) * CH; dst = won + (size_t)(r - O3) * CH; }
  int t = threadIdx.x;
  float a = src[t], b = src[t + 256];
  red[t] = a * a + b * b;
  __syncthreads();
  for (int s = 128; s > 0; s >>= 1) {
    if (t < s) red[t] += red[t + s];
    __syncthreads();
  }
  float scale = 1.0f / (sqrtf(red[0]) + 1e-4f * 22.627416997969522f);
  dst[t]       = (__bf16)(a * scale);
  dst[t + 256] = (__bf16)(b * scale);
}

// ---------------------------------------------------------------------------
// Kernel 2: mem-KV rows (pixel-normed) + zero padding rows for K/V buffers
// ---------------------------------------------------------------------------
__global__ void __launch_bounds__(64)
kvmem_kernel(const float* __restrict__ mem_kv,
             __bf16* __restrict__ kbuf, __bf16* __restrict__ vbuf) {
  int bh = blockIdx.x;            // 0..127  (b*8 + h)
  int h = bh & 7;
  int d = threadIdx.x;            // 0..63
  for (int kv = 0; kv < 2; ++kv) {
    __bf16* dst = kv ? vbuf : kbuf;
    for (int r = 0; r < 4; ++r) {
      const float* row = mem_kv + (((size_t)kv * NH + h) * 4 + r) * DH;
      float ss = 0.f;
      for (int e = 0; e < DH; ++e) ss += row[e] * row[e];
      float sc = 8.0f / fmaxf(sqrtf(ss), 1e-4f);
      dst[((size_t)bh * NKVP + r) * DH + d] = (__bf16)(row[d] * sc);
    }
    for (int r = NKV; r < NKVP; ++r)
      dst[((size_t)bh * NKVP + r) * DH + d] = (__bf16)0.0f;
  }
}

// ---------------------------------------------------------------------------
// Kernel 3: QKV GEMM. One wave -> 16 pixels x 64 chans (one head of q|k|v).
// K-loop over 512 in steps of 32, fused pixel-norm epilogue, bf16 output.
// ---------------------------------------------------------------------------
__global__ void __launch_bounds__(256)
qkv_gemm_kernel(const float* __restrict__ x, const __bf16* __restrict__ wqn,
                __bf16* __restrict__ qb, __bf16* __restrict__ kbuf,
                __bf16* __restrict__ vbuf) {
  int tid  = threadIdx.x;
  int lane = tid & 31;
  int half = lane >> 4;
  int nl   = lane & 15;
  int wave = blockIdx.x * 8 + (tid >> 5);
  int rt    = wave / 24;          // 16-pixel row tile, 0..1023
  int g     = wave % 24;          // (which, head)
  int which = g >> 3;
  int head  = g & 7;
  int obase = which * 512 + head * 64;

  v8f acc[4];
#pragma unroll
  for (int t = 0; t < 4; ++t) acc[t] = (v8f)0.0f;

  int pg = rt * 16 + nl;          // pixel row for A loads
  int bb = pg >> 10;
  const float* xbase = x + (size_t)bb * CH * NPIX + (pg & 1023);

  for (int c0 = 0; c0 < CH; c0 += 32) {
    BFV A;
#pragma unroll
    for (int j = 0; j < 8; ++j) {
      int kk = a_kidx(j, half);
      float a0 = xbase[(size_t)(c0 + kk) * NPIX];
      float a1 = xbase[(size_t)(c0 + kk + 1) * NPIX];
      A.u[j] = pack_f2bf(a0, a1);
    }
#pragma unroll
    for (int t = 0; t < 4; ++t) {
      int o = obase + t * 16 + nl;
      const unsigned* wp = (const unsigned*)(wqn + (size_t)o * CH + c0 + half * 16);
      BFV Bm;
#pragma unroll
      for (int j = 0; j < 8; ++j) Bm.u[j] = wp[j];
      acc[t] = __builtin_amdgcn_wmma_f32_16x16x32_bf16(
          false, A.v, false, Bm.v, (short)0, acc[t], false, false);
    }
  }

  // fused pixel_norm: t / max(||t||, eps) * sqrt(64)
  int bh = bb * 8 + head;
#pragma unroll
  for (int i = 0; i < 8; ++i) {
    float ss = 0.f;
#pragma unroll
    for (int t = 0; t < 4; ++t) ss += acc[t][i] * acc[t][i];
    ss = red_sum16(ss);
    float sc = 8.0f / fmaxf(sqrtf(ss), 1e-4f);
    int mr   = i + 8 * half;
    int prow = (rt * 16 + mr) & 1023;
#pragma unroll
    for (int t = 0; t < 4; ++t) {
      float v = acc[t][i] * sc;
      int d = t * 16 + nl;
      if (which == 0)
        qb[((size_t)bh * NPIX + prow) * DH + d] = (__bf16)v;
      else if (which == 1)
        kbuf[((size_t)bh * NKVP + 4 + prow) * DH + d] = (__bf16)v;
      else
        vbuf[((size_t)bh * NKVP + 4 + prow) * DH + d] = (__bf16)v;
    }
  }
}

// ---------------------------------------------------------------------------
// Kernel 4: flash attention. 4 waves/block share K/V LDS tiles (32 keys x 64),
// loaded via the Tensor Data Mover (TENSORcnt) when available. Each wave owns
// 16 query rows; S and P·V through WMMA; online softmax.
// ---------------------------------------------------------------------------
__global__ void __launch_bounds__(128)
attn_kernel(const __bf16* __restrict__ qb, const __bf16* __restrict__ kbuf,
            const __bf16* __restrict__ vbuf, __bf16* __restrict__ obuf) {
  __shared__ __attribute__((aligned(16))) __bf16 Kl[32 * 64];
  __shared__ __attribute__((aligned(16))) __bf16 Vl[32 * 64];
  __shared__ __attribute__((aligned(16))) __bf16 Pl[4 * 16 * 32];

  int tid  = threadIdx.x;
  int lane = tid & 31;
  int half = lane >> 4;
  int nl   = lane & 15;
  int ww   = tid >> 5;
  int bh   = blockIdx.x >> 4;      // 0..127
  int qtile = (blockIdx.x & 15) * 4 + ww;   // 0..63

  // Q tile (16 x 64) in A layout: two 16x32 operand slabs
  BFV aq[2];
  {
    const __bf16* qrow = qb + ((size_t)bh * NPIX + qtile * 16 + nl) * DH;
#pragma unroll
    for (int t = 0; t < 2; ++t)
#pragma unroll
      for (int j = 0; j < 8; ++j)
        aq[t].u[j] = *(const unsigned*)(qrow + t * 32 + a_kidx(j, half));
  }

  float mrow[8], lrow[8];
#pragma unroll
  for (int i = 0; i < 8; ++i) { mrow[i] = -3.0e38f; lrow[i] = 0.f; }
  v8f acc[4];
#pragma unroll
  for (int g2 = 0; g2 < 4; ++g2) acc[g2] = (v8f)0.0f;

  const __bf16* kbase = kbuf + (size_t)bh * NKVP * DH;
  const __bf16* vbase = vbuf + (size_t)bh * NKVP * DH;
  __bf16* pp = Pl + ww * (16 * 32);

  for (int jc = 0; jc < 33; ++jc) {
    __syncthreads();               // prior-iteration LDS readers done
#if USE_TDM
    if (ww == 0) {                 // one wave per block drives the TDM
      tdm_load_tile_bf16(kbase + (size_t)jc * 32 * DH, Kl, 32);
      tdm_load_tile_bf16(vbase + (size_t)jc * 32 * DH, Vl, 32);
      __builtin_amdgcn_s_wait_tensorcnt(0);
    }
#else
    {
      const unsigned* kgl = (const unsigned*)kbase;
      const unsigned* vgl = (const unsigned*)vbase;
      unsigned* kls = (unsigned*)Kl;
      unsigned* vls = (unsigned*)Vl;
      for (int i2 = tid; i2 < 1024; i2 += 128) {
        kls[i2] = kgl[jc * 1024 + i2];
        vls[i2] = vgl[jc * 1024 + i2];
      }
    }
#endif
    __syncthreads();               // tiles visible to all 4 waves

    // S = Q K^T for two 16-key halves; K-dim 64 = 2 WMMA steps each
    v8f s0 = (v8f)0.0f, s1 = (v8f)0.0f;
#pragma unroll
    for (int t = 0; t < 2; ++t) {
      const unsigned* kp0 = (const unsigned*)(Kl + (size_t)nl * DH + t * 32 + half * 16);
      const unsigned* kp1 = (const unsigned*)(Kl + (size_t)(16 + nl) * DH + t * 32 + half * 16);
      BFV B0, B1;
#pragma unroll
      for (int j = 0; j < 8; ++j) { B0.u[j] = kp0[j]; B1.u[j] = kp1[j]; }
      s0 = __builtin_amdgcn_wmma_f32_16x16x32_bf16(false, aq[t].v, false, B0.v, (short)0, s0, false, false);
      s1 = __builtin_amdgcn_wmma_f32_16x16x32_bf16(false, aq[t].v, false, B1.v, (short)0, s1, false, false);
    }

    int j0 = jc * 32 + nl;
    int j1 = j0 + 16;
#pragma unroll
    for (int i = 0; i < 8; ++i) {
      float a0 = s0[i] * 0.125f;
      float a1 = s1[i] * 0.125f;
      if (j0 >= NKV) a0 = -3.0e38f;   // mask padded keys
      if (j1 >= NKV) a1 = -3.0e38f;
      float rm   = red_max16(fmaxf(a0, a1));
      float mnew = fmaxf(mrow[i], rm);
      float alpha = __expf(mrow[i] - mnew);
      float p0 = __expf(a0 - mnew);
      float p1 = __expf(a1 - mnew);
      float rs = red_sum16(p0 + p1);
      lrow[i] = lrow[i] * alpha + rs;
      mrow[i] = mnew;
#pragma unroll
      for (int g2 = 0; g2 < 4; ++g2) acc[g2][i] *= alpha;
      int mr = i + 8 * half;
      pp[mr * 32 + nl]      = (__bf16)p0;   // stage P (C layout -> row major)
      pp[mr * 32 + 16 + nl] = (__bf16)p1;
    }
    asm volatile("s_wait_dscnt 0" ::: "memory");   // wave-private LDS RAW

    // reload P in A layout
    BFV pa;
#pragma unroll
    for (int j = 0; j < 8; ++j)
      pa.u[j] = *(const unsigned*)(pp + nl * 32 + a_kidx(j, half));

    // O += P V   (K-dim = 32 keys, one WMMA per 16-wide d group)
#pragma unroll
    for (int g2 = 0; g2 < 4; ++g2) {
      BFV Bv;
      int d = g2 * 16 + nl;
#pragma unroll
      for (int j = 0; j < 8; ++j) {
        int key = half * 16 + 2 * j;
        Bv.u[j] = pack_bf2(Vl[(size_t)key * DH + d], Vl[(size_t)(key + 1) * DH + d]);
      }
      acc[g2] = __builtin_amdgcn_wmma_f32_16x16x32_bf16(false, pa.v, false, Bv.v, (short)0, acc[g2], false, false);
    }
  }

  // epilogue: O /= l, write bf16 pixel-major [p, head*64+d] for the out GEMM
  int b = bh >> 3, h = bh & 7;
#pragma unroll
  for (int i = 0; i < 8; ++i) {
    float inv = 1.0f / lrow[i];
    int mr = i + 8 * half;
    size_t rowoff = ((size_t)b * NPIX + qtile * 16 + mr) * CH + h * DH;
#pragma unroll
    for (int g2 = 0; g2 < 4; ++g2)
      obuf[rowoff + g2 * 16 + nl] = (__bf16)(acc[g2][i] * inv);
  }
}

// ---------------------------------------------------------------------------
// Kernel 5: output projection GEMM + fused MPAdd residual, fp32 NCHW output
// ---------------------------------------------------------------------------
__global__ void __launch_bounds__(256)
out_gemm_kernel(const __bf16* __restrict__ obuf, const __bf16* __restrict__ won,
                const float* __restrict__ x, float* __restrict__ out) {
  int tid  = threadIdx.x;
  int lane = tid & 31;
  int half = lane >> 4;
  int nl   = lane & 15;
  int wave = blockIdx.x * 8 + (tid >> 5);
  int rt = wave >> 3;             // 16-pixel row tile, 0..1023
  int g  = wave & 7;              // 64-wide output channel group
  int obase = g * 64;

  v8f acc[4];
#pragma unroll
  for (int t = 0; t < 4; ++t) acc[t] = (v8f)0.0f;

  const __bf16* arow = obuf + (size_t)(rt * 16 + nl) * CH;

  for (int c0 = 0; c0 < CH; c0 += 32) {
    BFV A;
#pragma unroll
    for (int j = 0; j < 8; ++j)
      A.u[j] = *(const unsigned*)(arow + c0 + a_kidx(j, half));
#pragma unroll
    for (int t = 0; t < 4; ++t) {
      int o = obase + t * 16 + nl;
      const unsigned* wp = (const unsigned*)(won + (size_t)o * CH + c0 + half * 16);
      BFV Bm;
#pragma unroll
      for (int j = 0; j < 8; ++j) Bm.u[j] = wp[j];
      acc[t] = __builtin_amdgcn_wmma_f32_16x16x32_bf16(
          false, A.v, false, Bm.v, (short)0, acc[t], false, false);
    }
  }

  const float tres = 0.3f;
  const float inv_den = 1.3130643285972254f;   // 1/sqrt(0.7^2 + 0.3^2)
#pragma unroll
  for (int i = 0; i < 8; ++i) {
    int mr = i + 8 * half;
    int pg = rt * 16 + mr;
    int b = pg >> 10, p = pg & 1023;
#pragma unroll
    for (int t = 0; t < 4; ++t) {
      int o = obase + t * 16 + nl;
      size_t idx = ((size_t)b * CH + o) * NPIX + p;
      out[idx] = (acc[t][i] * (1.0f - tres) + x[idx] * tres) * inv_den;
    }
  }
}

// ---------------------------------------------------------------------------
extern "C" void kernel_launch(void* const* d_in, const int* in_sizes, int n_in,
                              void* d_out, int out_size, void* d_ws, size_t ws_size,
                              hipStream_t stream) {
  (void)in_sizes; (void)n_in; (void)out_size; (void)ws_size;
  const float* x      = (const float*)d_in[0];
  const float* w_qkv  = (const float*)d_in[1];
  const float* w_out  = (const float*)d_in[2];
  const float* mem_kv = (const float*)d_in[3];
  float* out = (float*)d_out;

  char* p = (char*)d_ws;
  auto carve = [&](size_t bytes) {
    char* r = p;
    p += (bytes + 255) & ~(size_t)255;
    return r;
  };
  __bf16* wqn  = (__bf16*)carve((size_t)O3 * CH * 2);
  __bf16* won  = (__bf16*)carve((size_t)CH * CH * 2);
  __bf16* qb   = (__bf16*)carve((size_t)16 * NH * NPIX * DH * 2);
  __bf16* kbuf = (__bf16*)carve((size_t)16 * NH * NKVP * DH * 2);
  __bf16* vbuf = (__bf16*)carve((size_t)16 * NH * NKVP * DH * 2);
  __bf16* obuf = (__bf16*)carve((size_t)16 * NPIX * CH * 2);

  hipLaunchKernelGGL(wnorm_kernel, dim3(2048), dim3(256), 0, stream, w_qkv, w_out, wqn, won);
  hipLaunchKernelGGL(kvmem_kernel, dim3(128), dim3(64), 0, stream, mem_kv, kbuf, vbuf);
  hipLaunchKernelGGL(qkv_gemm_kernel, dim3(3072), dim3(256), 0, stream, x, wqn, qb, kbuf, vbuf);
  hipLaunchKernelGGL(attn_kernel, dim3(2048), dim3(128), 0, stream, qb, kbuf, vbuf, obuf);
  hipLaunchKernelGGL(out_gemm_kernel, dim3(1024), dim3(256), 0, stream, obuf, won, x, out);
}
